// IIA_38414187495584
// MI455X (gfx1250) — compile-verified, matrix operand order checked by reading, not run on hardware
//
#include <hip/hip_runtime.h>
#include <hip/hip_bf16.h>

// ---------------------------------------------------------------------------
// Problem constants (from reference): B=16, H=W=512, CNUM=6, LNUM=3, AUG sel=0
// ---------------------------------------------------------------------------
#define BB   16
#define HH   512
#define WW   512
#define CNUMK 6
#define LNUMK 3

typedef __attribute__((ext_vector_type(16))) _Float16 v16h;
typedef __attribute__((ext_vector_type(8)))  _Float16 v8h;
typedef __attribute__((ext_vector_type(8)))  float    v8f;

__device__ __forceinline__ float sigmoidf(float x) {
    return 1.0f / (1.0f + __expf(-x));
}

// K-index of element e of the v16h operand held by `lane` for 16-bit WMMA
// A/B operands (CDNA5 ISA 7.12.2):
//   lanes 0-15 : elements 0..7 -> K 0..7,  elements 8..15 -> K 16..23
//   lanes 16-31: elements 0..7 -> K 8..15, elements 8..15 -> K 24..31
__device__ __forceinline__ int koff16(int lane, int e) {
    int base = (e < 8) ? e : (e + 8);
    return (lane < 16) ? base : (base + 8);
}

// ---------------------------------------------------------------------------
// Per-sample min / (max-min+eps)
// ---------------------------------------------------------------------------
__global__ __launch_bounds__(256)
void minmax_kernel(const float* __restrict__ x, float* __restrict__ mm) {
    __shared__ float smn[256], smx[256];
    const int b = blockIdx.x;
    const float* p = x + (size_t)b * (HH * WW);
    float mn = 3.4e38f, mx = -3.4e38f;
    for (int i = threadIdx.x; i < HH * WW; i += 256) {
        float v = p[i];
        mn = fminf(mn, v);
        mx = fmaxf(mx, v);
    }
    smn[threadIdx.x] = mn; smx[threadIdx.x] = mx;
    __syncthreads();
    for (int off = 128; off > 0; off >>= 1) {
        if (threadIdx.x < off) {
            smn[threadIdx.x] = fminf(smn[threadIdx.x], smn[threadIdx.x + off]);
            smx[threadIdx.x] = fmaxf(smx[threadIdx.x], smx[threadIdx.x + off]);
        }
        __syncthreads();
    }
    if (threadIdx.x == 0) {
        mm[b]      = smn[0];
        mm[BB + b] = smx[0] - smn[0] + 1e-8f;  // scale
    }
}

__global__ __launch_bounds__(256)
void norm_kernel(const float* __restrict__ x, const float* __restrict__ mm,
                 float* __restrict__ xn) {
    int i = blockIdx.x * 256 + threadIdx.x;
    if (i < BB * HH * WW) {
        int b = i >> 18;                    // 512*512 = 2^18
        xn[i] = (x[i] - mm[b]) / mm[BB + b];
    }
}

__global__ __launch_bounds__(256)
void denorm_kernel(const float* __restrict__ xn, const float* __restrict__ mm,
                   float* __restrict__ out) {
    int i = blockIdx.x * 256 + threadIdx.x;
    if (i < BB * HH * WW) {
        int b = i >> 18;
        out[i] = xn[i] * mm[BB + b] + mm[b];
    }
}

// ---------------------------------------------------------------------------
// Fused per-class kernel: conv1(1->4) x3 layers + conv2(4->1) x3 layers via
// WMMA im2col GEMMs entirely in LDS, then 3-step Bezier blend, masked write.
// One block = one 16x16 output tile of one sample. 256 threads = 8 wave32.
//
// Key tricks:
//  * any K index outside the real reduction range (tap>=9 in conv1, kg>=108
//    in conv2) has an all-zero B row, so the matching A elements may be
//    garbage -> all A gathers are unconditional (no exec-masked loads).
//  * B operands are materialized as ZERO-PADDED, pre-diagonalized f16 tables
//    in LDS -> each lane's B fragment is two aligned 16-byte ds_load_b128s,
//    with no compares/selects at the WMMA sites.
// ---------------------------------------------------------------------------
#define Y1STRIDE 344   // halves per channel row; *2 bytes = 688 (16B aligned)

__global__ __launch_bounds__(256)
void class_kernel(const float* __restrict__ xn_in, float* __restrict__ xn_out,
                  const int* __restrict__ lbl, const int* __restrict__ index,
                  const float* __restrict__ param, const float* __restrict__ w1g,
                  const float* __restrict__ b1g, const float* __restrict__ w2g,
                  const float* __restrict__ b2g, int cls)
{
    __shared__ float    rawT[20][20];                       // input tile + 2-halo
    __shared__ __align__(16) _Float16 y1T[12 * Y1STRIDE];   // conv1 out, 12ch x 18x18
    __shared__ __align__(16) _Float16 bmat1T[16 * 32];      // conv1 B: [col][K], 0-padded
    __shared__ __align__(16) _Float16 b2T[16 * 128];        // conv2 B: [col][K], block-diag
    __shared__ float    mixT[3][258];                       // sigmoid(conv2), 3 layers
    __shared__ float    sW1[108];                           // [layer][oc][tap]
    __shared__ float    sW2[108];                           // [layer][ic][tap]
    __shared__ float    sB1[12];
    __shared__ float    sB2[3];
    __shared__ float    sPP[3][4];                          // sigmoid(param) per layer

    const int tid  = threadIdx.x;
    const int lane = tid & 31;
    const int wv   = tid >> 5;
    const int b    = blockIdx.z;
    const int x0   = blockIdx.x * 16;
    const int y0   = blockIdx.y * 16;
    const int idx  = index[b];
    const int ic6  = idx * CNUMK + cls;   // (index, class, aug=0) group

    // Warm the label tile lines we will read at the end.
    __builtin_prefetch(&lbl[((size_t)b * HH + y0) * WW + x0], 0, 1);

    // Per-(sample,class) weights -> LDS.
    // w1: (64,6,4,3,4,1,3,3) -> per (idx,cls): 432 at aug 0; same for w2.
    if (tid < 108) {
        sW1[tid] = w1g[(size_t)ic6 * 432 + tid];
        sW2[tid] = w2g[(size_t)ic6 * 432 + tid];
    }
    if (tid >= 128 && tid < 140) sB1[tid - 128] = b1g[ic6 * 48 + (tid - 128)];
    if (tid >= 144 && tid < 147) sB2[tid - 144] = b2g[ic6 * 12 + (tid - 144)];
    if (tid >= 160 && tid < 172) {
        int l = (tid - 160) >> 2, s = (tid - 160) & 3;
        sPP[l][s] = sigmoidf(param[ic6 * 84 + l * 7 + s]);
    }

    // Input tile with 2-pixel halo (SAME conv pads with zeros).
    for (int k = tid; k < 400; k += 256) {
        int ry = k / 20, rx = k % 20;
        int gy = y0 + ry - 2, gx = x0 + rx - 2;
        float v = 0.0f;
        if (gy >= 0 && gy < HH && gx >= 0 && gx < WW)
            v = xn_in[((size_t)b * HH + gy) * WW + gx];
        rawT[ry][rx] = v;
    }
    __syncthreads();

    // Build zero-padded f16 B tables (branch-free: clamped load + select).
    // conv1 B: column n = layer*4+oc (cols 12..15 zero), K = tap (9..31 zero).
    for (int j = tid; j < 16 * 32; j += 256) {
        int col = j >> 5, k = j & 31;
        int cc2 = (col < 12) ? col : 11;
        int kc  = (k < 9) ? k : 8;
        float wraw = sW1[(cc2 >> 2) * 36 + (cc2 & 3) * 9 + kc];
        bmat1T[j] = (_Float16)(((col < 12) && (k < 9)) ? wraw : 0.0f);
    }
    // conv2 B: column n = layer (block-diagonal over K = layer*36+ic*9+tap).
    for (int j = tid; j < 16 * 128; j += 256) {
        int col = j >> 7, k = j & 127;
        int kc  = (k < 108) ? k : 107;
        float wraw = sW2[kc];
        b2T[j] = (_Float16)(((k < 108) && (col == kc / 36)) ? wraw : 0.0f);
    }
    __syncthreads();

    const int  n      = lane & 15;         // WMMA column this lane owns
    const bool hiHalf = (lane >= 16);
    const int  hb8    = hiHalf ? 8 : 0;    // K sub-offset of this half-wave

    // ---- conv1: D[16px x 12] = A[16px x 9taps] * W1[9 x 12] + bias ----------
    v16h bmat1;
    {
        v8h lo = *(const v8h*)&bmat1T[n * 32 + hb8];        // K chunk 0..15
        v8h hi = *(const v8h*)&bmat1T[n * 32 + 16 + hb8];   // K chunk 16..31
        bmat1 = __builtin_shufflevector(lo, hi, 0, 1, 2, 3, 4, 5, 6, 7,
                                        8, 9, 10, 11, 12, 13, 14, 15);
    }
    float bias1 = (n < 12) ? sB1[n] : 0.0f;
    v8f cinit1;
    #pragma unroll
    for (int r = 0; r < 8; ++r) cinit1[r] = bias1;

    // y1 needed on 18x18 = 324 pixels -> 21 strips of 16, round-robin on waves
    for (int s = wv; s < 21; s += 8) {
        int p  = s * 16 + n;
        int pc = (p < 324) ? p : 323;     // clamp tail strip (stores are in-bounds pad)
        int py = pc / 18, px = pc % 18;
        v16h amat;
        // Element 0: tap 0 (lanes 0-15) or tap 8 (lanes 16-31) -> one select.
        {
            int dy = hiHalf ? 2 : 0, dx = hiHalf ? 2 : 0;
            amat[0] = (_Float16)rawT[py + dy][px + dx];
        }
        // Elements 1..7: taps 1..7 for lanes 0-15; lanes 16-31 hit zero B rows.
        #pragma unroll
        for (int e = 1; e < 8; ++e)
            amat[e] = (_Float16)rawT[py + e / 3][px + e % 3];
        // Elements 8..15: K >= 16 -> zero B rows; value irrelevant.
        #pragma unroll
        for (int e = 8; e < 16; ++e) amat[e] = (_Float16)0.0f;

        v8f d = __builtin_amdgcn_wmma_f32_16x16x32_f16(
            false, amat, false, bmat1, (short)0, cinit1, false, false);

        // Packed 16-byte scatter: lane's 8 rows are 8 consecutive pixels.
        if (n < 12) {
            v8h hv;
            #pragma unroll
            for (int r = 0; r < 8; ++r) hv[r] = (_Float16)d[r];
            int pbase = s * 16 + hb8;                  // <= 335 < Y1STRIDE
            *(v8h*)&y1T[n * Y1STRIDE + pbase] = hv;
        }
    }
    __syncthreads();

    // ---- conv2: D[16px x 3layers] = A[16px x 108] * blockdiag(W2)[108 x 3] --
    // B chunks: pure aligned 16B table loads, strip-invariant.
    v16h bch[4];
    #pragma unroll
    for (int cc = 0; cc < 4; ++cc) {
        v8h lo = *(const v8h*)&b2T[n * 128 + cc * 32 + hb8];
        v8h hi = *(const v8h*)&b2T[n * 128 + cc * 32 + 16 + hb8];
        bch[cc] = __builtin_shufflevector(lo, hi, 0, 1, 2, 3, 4, 5, 6, 7,
                                          8, 9, 10, 11, 12, 13, 14, 15);
    }
    float bias2 = (n < 3) ? sB2[n] : 0.0f;
    v8f cinit2;
    #pragma unroll
    for (int r = 0; r < 8; ++r) cinit2[r] = bias2;

    for (int s2 = wv; s2 < 16; s2 += 8) {   // 256 output px -> 16 strips
        int q  = s2 * 16 + n;
        int oy = q >> 4, ox = q & 15;
        v8f acc = cinit2;
        #pragma unroll
        for (int cc = 0; cc < 4; ++cc) {
            v16h a2;
            #pragma unroll
            for (int e = 0; e < 16; ++e) {
                int kg  = cc * 32 + koff16(lane, e);
                int kgc = (kg < 108) ? kg : 107;     // clamp: kg>=108 rows of B are 0
                int l = kgc / 36, r2 = kgc % 36;
                int chn = r2 / 9, t = r2 % 9;
                a2[e] = y1T[(l * 4 + chn) * Y1STRIDE + (oy + t / 3) * 18 + (ox + t % 3)];
            }
            acc = __builtin_amdgcn_wmma_f32_16x16x32_f16(
                false, a2, false, bch[cc], (short)0, acc, false, false);
        }
        if (n < 3) {
            #pragma unroll
            for (int r = 0; r < 8; ++r) {
                int mrow = r + hb8;
                mixT[n][s2 * 16 + mrow] = sigmoidf(acc[r]);
            }
        }
    }
    __syncthreads();

    // ---- Bezier blend, masked write-back -----------------------------------
    {
        int q  = tid;
        int oy = q >> 4, ox = q & 15;
        int gy = y0 + oy, gx = x0 + ox;
        float raw = rawT[oy + 2][ox + 2];
        int lab = lbl[((size_t)b * HH + gy) * WW + gx];
        bool msk = (lab == cls);
        float c = msk ? raw : 0.0f;
        #pragma unroll
        for (int l = 0; l < LNUMK; ++l) {
            float mixv = mixT[l][q];
            float p1 = sPP[l][0], p2 = sPP[l][1];
            float p1v = sPP[l][2], p2v = sPP[l][3];
            float om = 1.0f - c;
            float ct = 3.0f * om * om * c * p1 + 3.0f * om * c * c * p2 + c * c * c;
            float cv = om * om * om + 3.0f * om * om * c * p1v + 3.0f * om * c * c * p2v;
            c = fminf(fmaxf(ct * mixv + cv * (1.0f - mixv), 0.0f), 1.0f);
        }
        xn_out[((size_t)b * HH + gy) * WW + gx] = msk ? c : raw;
    }
}

// ---------------------------------------------------------------------------
extern "C" void kernel_launch(void* const* d_in, const int* in_sizes, int n_in,
                              void* d_out, int out_size, void* d_ws, size_t ws_size,
                              hipStream_t stream) {
    const float* x     = (const float*)d_in[0];  // GLA_img_aug [16,1,512,512]
    const int*   lbl   = (const int*)  d_in[1];  // [16,512,512]
    const int*   index = (const int*)  d_in[2];  // [16]
    const float* param = (const float*)d_in[3];  // [64,6,4,3,7]
    const float* w1    = (const float*)d_in[4];  // [64,6,4,3,4,1,3,3]
    const float* b1    = (const float*)d_in[5];  // [64,6,4,3,4]
    const float* w2    = (const float*)d_in[6];  // [64,6,4,3,1,4,3,3]
    const float* b2    = (const float*)d_in[7];  // [64,6,4,3,1]
    float*       out   = (float*)d_out;

    float* ws = (float*)d_ws;
    float* mm = ws;                                   // mn[16], scale[16]
    float* X0 = ws + 64;
    float* X1 = X0 + (size_t)BB * HH * WW;

    const int total   = BB * HH * WW;
    const int eblocks = (total + 255) / 256;

    minmax_kernel<<<BB, 256, 0, stream>>>(x, mm);
    norm_kernel<<<eblocks, 256, 0, stream>>>(x, mm, X0);

    float* cur = X0; float* nxt = X1;
    for (int cls = 0; cls < CNUMK; ++cls) {
        class_kernel<<<dim3(WW / 16, HH / 16, BB), 256, 0, stream>>>(
            cur, nxt, lbl, index, param, w1, b1, w2, b2, cls);
        float* t = cur; cur = nxt; nxt = t;
    }
    // 6 swaps -> result back in X0 (== cur)
    denorm_kernel<<<eblocks, 256, 0, stream>>>(cur, mm, out);
}
